// Decoder_8186207666796
// MI455X (gfx1250) — compile-verified, matrix operand order checked by reading, not run on hardware
//
#include <hip/hip_runtime.h>

// ---------------------------------------------------------------------------
// Types for CDNA5 WMMA (wave32): v_wmma_f32_16x16x32_bf16
// ---------------------------------------------------------------------------
typedef __attribute__((ext_vector_type(16))) __bf16          v16bf;
typedef __attribute__((ext_vector_type(8)))  float           v8f;
typedef __attribute__((ext_vector_type(8)))  unsigned short  v8us;

union FragU { v16bf bf; v8us us[2]; };

__device__ __forceinline__ unsigned short f2bf(float f) {
  union { float f; unsigned int u; } c; c.f = f;
  unsigned int u = c.u;
  unsigned int r = u + 0x7fffu + ((u >> 16) & 1u);  // round-to-nearest-even
  return (unsigned short)(r >> 16);
}

// Load one 16-bit WMMA operand fragment: lane holds row (lane&15),
// K elements [kb..kb+7] and [kb+16..kb+23], kb=(lane>>4)*8.  Caller passes a
// pointer already offset to (row, kb); we fetch 8 halves at +0 and +16.
__device__ __forceinline__ v16bf load_frag(const unsigned short* p) {
  FragU f;
  f.us[0] = *(const v8us*)(p);
  f.us[1] = *(const v8us*)(p + 16);
  return f.bf;
}

__device__ __forceinline__ v8f wmma_bf16(v16bf a, v16bf b, v8f c) {
  return __builtin_amdgcn_wmma_f32_16x16x32_bf16(false, a, false, b,
                                                 (short)0, c, false, false);
}

// ---------------------------------------------------------------------------
// Generic batched bf16 GEMM:  C[z] = A[z] (MxK) * Bt[z]^T (Bt is NxK) + bias
// z decomposed as (zb, zh) = (z/Hdim, z%Hdim) with separate strides, which
// covers flat projections (Hdim=1) and per-(batch,head) attention GEMMs.
//
// Workgroup = 8 waves; each wave computes a 32(M) x 64(N) tile -> 256x64 per
// block.  The 64x32-half B-panel is shared by all 8 waves, so it is staged
// once per block into LDS with CDNA5 async global->LDS copies
// (GLOBAL_LOAD_ASYNC_TO_LDS_B128, ASYNCcnt), double-buffered so the DMA of
// K-step i+1 overlaps the WMMAs of step i.  One s_wait_asynccnt + barrier
// per step.  A-fragments stream straight from global (no cross-wave reuse).
//
// Requires: M%256==0 (per z), N%64==0, K%32==0, operands 16B aligned.
// ---------------------------------------------------------------------------
#define PANEL_H 64   // N rows in panel
#define PANEL_W 40   // 32 K-halves + 8 pad halves -> 80B row stride (16B mult)

__global__ __launch_bounds__(256) void gemm_bf16_wmma(
    const unsigned short* __restrict__ A,  long lda,
    const unsigned short* __restrict__ Bt, long ldb,
    const float* __restrict__ bias,
    float* __restrict__ Cf, unsigned short* __restrict__ Cb, long ldc,
    int M, int N, int K, int relu,
    long sAb, long sAh, long sBb, long sBh, long sCb, long sCh, int Hdim)
{
  __shared__ unsigned short panel[2][PANEL_H * PANEL_W];

  const int z  = blockIdx.z;
  const int zb = z / Hdim, zh = z - zb * Hdim;
  A  += (long)zb * sAb + (long)zh * sAh;
  Bt += (long)zb * sBb + (long)zh * sBh;
  const long coff = (long)zb * sCb + (long)zh * sCh;

  const int tid   = threadIdx.x;
  const int wave  = tid >> 5;
  const int lane  = tid & 31;
  const int lhalf = lane >> 4;        // 0 or 1
  const int l16   = lane & 15;

  const int m0 = blockIdx.y * 256 + wave * 32;
  const int n0 = blockIdx.x * 64;

  // --- async staging addresses: thread -> (row, 16B chunk) of 64x32 panel
  const int srow = tid >> 2, schunk = tid & 3;
  const unsigned short* gB = Bt + (long)(n0 + srow) * ldb + schunk * 8;
  const unsigned ldsA =
      (unsigned)(uintptr_t)(&panel[0][srow * PANEL_W]) + (unsigned)(schunk * 16);
  const unsigned ldsB = ldsA + (unsigned)(PANEL_H * PANEL_W * 2);

  const unsigned short* arow0 = A + (long)(m0 + l16) * lda + lhalf * 8;
  const unsigned short* arow1 = arow0 + 16 * lda;
  const unsigned short* brow  = &panel[0][0] + l16 * PANEL_W + lhalf * 8;

  v8f acc[2][4];
#pragma unroll
  for (int i = 0; i < 2; ++i)
#pragma unroll
    for (int j = 0; j < 4; ++j) acc[i][j] = v8f{};

  // prologue: stage K-chunk 0 into buffer 0
  asm volatile("global_load_async_to_lds_b128 %0, %1, off"
               :: "v"(ldsA), "v"((unsigned long long)(uintptr_t)gB)
               : "memory");

  const int nsteps = K >> 5;
  for (int it = 0; it < nsteps; ++it) {
    const int cur = it & 1;
    // own async writes retired; barrier => all waves' writes visible and all
    // prior-step reads of the other buffer finished
    asm volatile("s_wait_asynccnt 0x0" ::: "memory");
    __syncthreads();
    if (it + 1 < nsteps) {  // stage next K-chunk into the other buffer
      const unsigned long long gnext =
          (unsigned long long)(uintptr_t)(gB + (long)(it + 1) * 32);
      const unsigned lnext = cur ? ldsA : ldsB;
      asm volatile("global_load_async_to_lds_b128 %0, %1, off"
                   :: "v"(lnext), "v"(gnext) : "memory");
    }
    const long k0 = (long)it * 32;
    __builtin_prefetch(arow0 + k0 + 64, 0, 3);       // global_prefetch_b8
    v16bf a0 = load_frag(arow0 + k0);
    v16bf a1 = load_frag(arow1 + k0);
    const unsigned short* bp = brow + (cur ? (PANEL_H * PANEL_W) : 0);
#pragma unroll
    for (int nt = 0; nt < 4; ++nt) {
      v16bf b = load_frag(bp + nt * 16 * PANEL_W);   // ds_load_b128 x2
      acc[0][nt] = wmma_bf16(a0, b, acc[0][nt]);
      acc[1][nt] = wmma_bf16(a1, b, acc[1][nt]);
    }
  }

  // C/D layout: VGPR r of lane -> row +r+8*lhalf, col n0 + nt*16 + l16
#pragma unroll
  for (int mt = 0; mt < 2; ++mt) {
#pragma unroll
    for (int nt = 0; nt < 4; ++nt) {
      const int n = n0 + nt * 16 + l16;
      const float bs = bias ? bias[n] : 0.0f;
#pragma unroll
      for (int r = 0; r < 8; ++r) {
        float v = acc[mt][nt][r] + bs;
        if (relu) v = fmaxf(v, 0.0f);
        const long idx = coff + (long)(m0 + mt * 16 + r + lhalf * 8) * ldc + n;
        if (Cf) Cf[idx] = v;
        if (Cb) Cb[idx] = f2bf(v);
      }
    }
  }
}

// ---------------------------------------------------------------------------
// Elementwise f32 -> bf16
// ---------------------------------------------------------------------------
__global__ void cvt_f32_bf16(const float* __restrict__ in,
                             unsigned short* __restrict__ out, long n) {
  long i = (long)blockIdx.x * blockDim.x + threadIdx.x;
  if (i < n) out[i] = f2bf(in[i]);
}

// ---------------------------------------------------------------------------
// Tiled transpose f32[R,C] -> bf16[C,R]   (weights, conv->kv); batched via z
// ---------------------------------------------------------------------------
__global__ void transpose_cvt(const float* __restrict__ in,
                              unsigned short* __restrict__ out,
                              int R, int C, long ldin, long ldout,
                              long sIb, long sOb) {
  __shared__ float t[32][33];
  const float* ip = in + (long)blockIdx.z * sIb;
  unsigned short* op = out + (long)blockIdx.z * sOb;
  const int r0 = blockIdx.y * 32, c0 = blockIdx.x * 32;
#pragma unroll
  for (int j = 0; j < 4; ++j)
    t[threadIdx.y + j * 8][threadIdx.x] =
        ip[(long)(r0 + threadIdx.y + j * 8) * ldin + c0 + threadIdx.x];
  __syncthreads();
#pragma unroll
  for (int j = 0; j < 4; ++j)
    op[(long)(c0 + threadIdx.y + j * 8) * ldout + r0 + threadIdx.x] =
        f2bf(t[threadIdx.x][threadIdx.y + j * 8]);
}

// bf16[R,C] (row stride ldin) -> bf16[C,R]; z decomposed as (b,h)
__global__ void transpose_b16(const unsigned short* __restrict__ in,
                              unsigned short* __restrict__ out,
                              int R, int C, long ldin, long ldout, int Hdim,
                              long sIb, long sIh, long sOb, long sOh) {
  __shared__ unsigned short t[32][33];
  const int z = blockIdx.z, zb = z / Hdim, zh = z - zb * Hdim;
  const unsigned short* ip = in + (long)zb * sIb + (long)zh * sIh;
  unsigned short* op = out + (long)zb * sOb + (long)zh * sOh;
  const int r0 = blockIdx.y * 32, c0 = blockIdx.x * 32;
#pragma unroll
  for (int j = 0; j < 4; ++j)
    t[threadIdx.y + j * 8][threadIdx.x] =
        ip[(long)(r0 + threadIdx.y + j * 8) * ldin + c0 + threadIdx.x];
  __syncthreads();
#pragma unroll
  for (int j = 0; j < 4; ++j)
    op[(long)(c0 + threadIdx.y + j * 8) * ldout + r0 + threadIdx.x] =
        t[threadIdx.x][threadIdx.y + j * 8];
}

// ---------------------------------------------------------------------------
// Row softmax over Tk (256 threads/row, Tk/256 <= 4 elements per thread).
// Applies scale, optional causal mask (col <= row%Tq allowed).
// Writes optional f32 probs (may alias input S: values are register-held)
// and bf16 probs.
// ---------------------------------------------------------------------------
__global__ __launch_bounds__(256) void softmax_rows(
    const float* __restrict__ S, float* __restrict__ Pf,
    unsigned short* __restrict__ Pb, int Tq, int Tk, float scale, int causal)
{
  __shared__ float red[256];
  const long row = blockIdx.x;
  const int q = (int)(row % Tq);
  const int tid = threadIdx.x;
  const int nper = Tk >> 8;
  const long base = row * (long)Tk;

  float v[4];
  float mx = -3.4e38f;
  for (int j = 0; j < nper; ++j) {
    const int c = tid + j * 256;
    float x = S[base + c] * scale;
    if (causal && c > q) x = -3.4e38f;
    v[j] = x;
    mx = fmaxf(mx, x);
  }
  red[tid] = mx; __syncthreads();
  for (int s = 128; s > 0; s >>= 1) {
    if (tid < s) red[tid] = fmaxf(red[tid], red[tid + s]);
    __syncthreads();
  }
  mx = red[0]; __syncthreads();

  float sum = 0.0f;
  for (int j = 0; j < nper; ++j) {
    const int c = tid + j * 256;
    float e = expf(v[j] - mx);
    if (causal && c > q) e = 0.0f;
    v[j] = e; sum += e;
  }
  red[tid] = sum; __syncthreads();
  for (int s = 128; s > 0; s >>= 1) {
    if (tid < s) red[tid] += red[tid + s];
    __syncthreads();
  }
  const float inv = 1.0f / red[0];

  for (int j = 0; j < nper; ++j) {
    const int c = tid + j * 256;
    const float p = v[j] * inv;
    if (Pf) Pf[base + c] = p;
    Pb[base + c] = f2bf(p);
  }
}

// ---------------------------------------------------------------------------
// out = g * (x+y - mean) / (std_bessel + eps) + b   over D=1024 rows.
// Optional bf16 copy. Of may alias X (values register-held).
// ---------------------------------------------------------------------------
__global__ __launch_bounds__(256) void add_layernorm(
    const float* __restrict__ X, const float* __restrict__ Y,
    const float* __restrict__ g, const float* __restrict__ bta,
    float* __restrict__ Of, unsigned short* __restrict__ Ob)
{
  __shared__ float red[256];
  const long base = (long)blockIdx.x * 1024;
  const int tid = threadIdx.x;

  float v[4], s = 0.0f;
#pragma unroll
  for (int j = 0; j < 4; ++j) {
    const int c = tid + j * 256;
    float x = X[base + c] + Y[base + c];
    v[j] = x; s += x;
  }
  red[tid] = s; __syncthreads();
  for (int st = 128; st > 0; st >>= 1) {
    if (tid < st) red[tid] += red[tid + st];
    __syncthreads();
  }
  const float mean = red[0] * (1.0f / 1024.0f); __syncthreads();

  float ss = 0.0f;
#pragma unroll
  for (int j = 0; j < 4; ++j) { const float d = v[j] - mean; ss += d * d; }
  red[tid] = ss; __syncthreads();
  for (int st = 128; st > 0; st >>= 1) {
    if (tid < st) red[tid] += red[tid + st];
    __syncthreads();
  }
  const float var = red[0] * (1.0f / 1023.0f);   // ddof=1 (torch .std default)
  const float inv = 1.0f / (sqrtf(var) + 1e-6f); // matches g*(x-m)/(s+eps)+b

#pragma unroll
  for (int j = 0; j < 4; ++j) {
    const int c = tid + j * 256;
    const float o = g[c] * (v[j] - mean) * inv + bta[c];
    Of[base + c] = o;
    if (Ob) Ob[base + c] = f2bf(o);
  }
}

// ---------------------------------------------------------------------------
// Host-side orchestration
// ---------------------------------------------------------------------------
extern "C" void kernel_launch(void* const* d_in, const int* in_sizes, int n_in,
                              void* d_out, int out_size, void* d_ws, size_t ws_size,
                              hipStream_t stream) {
  (void)in_sizes; (void)n_in; (void)out_size; (void)ws_size;
  constexpr int B = 16, T = 512, D = 1024, H = 16, DFF = 2048;
  constexpr int TKC = 1024;            // cross-attn kv tokens (32*32)
  constexpr long MQ  = (long)B * T;    // 8192 query rows
  constexpr long MKV = (long)B * TKC;  // 16384 kv rows

  const float* text = (const float*)d_in[0];
  const float* conv = (const float*)d_in[1];
  const float* sa_wq = (const float*)d_in[2];  const float* sa_bq = (const float*)d_in[3];
  const float* sa_wk = (const float*)d_in[4];  const float* sa_bk = (const float*)d_in[5];
  const float* sa_wv = (const float*)d_in[6];  const float* sa_bv = (const float*)d_in[7];
  const float* sa_wo = (const float*)d_in[8];  const float* sa_bo = (const float*)d_in[9];
  const float* ca_wq = (const float*)d_in[10]; const float* ca_bq = (const float*)d_in[11];
  const float* ca_wk = (const float*)d_in[12]; const float* ca_bk = (const float*)d_in[13];
  const float* ca_wv = (const float*)d_in[14]; const float* ca_bv = (const float*)d_in[15];
  const float* ca_wo = (const float*)d_in[16]; const float* ca_bo = (const float*)d_in[17];
  const float* ffn_w1 = (const float*)d_in[18]; const float* ffn_b1 = (const float*)d_in[19];
  const float* ffn_w2 = (const float*)d_in[20]; const float* ffn_b2 = (const float*)d_in[21];
  const float* ln1_g = (const float*)d_in[22]; const float* ln1_b = (const float*)d_in[23];
  const float* ln2_g = (const float*)d_in[24]; const float* ln2_b = (const float*)d_in[25];
  const float* ln3_g = (const float*)d_in[26]; const float* ln3_b = (const float*)d_in[27];

  float* outRes = (float*)d_out;                 // [B,T,D]
  float* attn   = outRes + MQ * D;               // [B,H,T,1024]; also score scratch

  // Workspace layout (MB offsets, ~552 MB total)
  const long MBy = 1024L * 1024L;
  unsigned char* ws = (unsigned char*)d_ws;
  unsigned short* WT[8];                          // sa q,k,v,o, ca q,k,v,o ([1024,1024] each)
  for (int i = 0; i < 8; ++i) WT[i] = (unsigned short*)(ws + (long)i * 2 * MBy);
  unsigned short* W1T = (unsigned short*)(ws + 16 * MBy);   // [2048,1024]
  unsigned short* W2T = (unsigned short*)(ws + 20 * MBy);   // [1024,2048]
  unsigned short* XB  = (unsigned short*)(ws + 24 * MBy);   // [8192,1024] bf16 activations
  unsigned short* QB  = (unsigned short*)(ws + 40 * MBy);   // [8192,1024]
  unsigned short* KB  = (unsigned short*)(ws + 56 * MBy);   // up to [16384,1024]
  unsigned short* VB  = (unsigned short*)(ws + 88 * MBy);   // up to [16384,1024]
  unsigned short* VT  = (unsigned short*)(ws + 120 * MBy);  // [B*H, 64, Tk]
  unsigned short* PB  = (unsigned short*)(ws + 152 * MBy);  // probs bf16, up to 256 MB
  unsigned short* AOB = (unsigned short*)(ws + 408 * MBy);  // [8192,1024] attn out bf16
  float*          PO  = (float*)        (ws + 424 * MBy);   // [8192,1024] proj out f32
  float*          RES = (float*)        (ws + 456 * MBy);   // [8192,1024] running result
  unsigned short* HB  = (unsigned short*)(ws + 488 * MBy);  // [8192,2048] ffn hidden bf16
  unsigned short* KVB = (unsigned short*)(ws + 520 * MBy);  // [16384,1024] kv tokens bf16

  const dim3 tb(32, 8, 1);
  const float* wsrc[8] = {sa_wq, sa_wk, sa_wv, sa_wo, ca_wq, ca_wk, ca_wv, ca_wo};

  // ---- 0. Transpose+convert all weights to bf16 [N,K] -------------------
  for (int i = 0; i < 8; ++i)
    transpose_cvt<<<dim3(32, 32, 1), tb, 0, stream>>>(wsrc[i], WT[i], 1024, 1024, 1024, 1024, 0, 0);
  transpose_cvt<<<dim3(64, 32, 1), tb, 0, stream>>>(ffn_w1, W1T, 1024, 2048, 2048, 1024, 0, 0);
  transpose_cvt<<<dim3(32, 64, 1), tb, 0, stream>>>(ffn_w2, W2T, 2048, 1024, 1024, 2048, 0, 0);

  // ---- 1. text -> bf16 --------------------------------------------------
  cvt_f32_bf16<<<(unsigned)((MQ * D) / 256), 256, 0, stream>>>(text, XB, MQ * D);

  // ---- 2. Self-attention Q/K/V projections (bf16 out) -------------------
  gemm_bf16_wmma<<<dim3(16, 32, 1), 256, 0, stream>>>(XB, 1024, WT[0], 1024, sa_bq,
      nullptr, QB, 1024, (int)MQ, 1024, 1024, 0, 0, 0, 0, 0, 0, 0, 1);
  gemm_bf16_wmma<<<dim3(16, 32, 1), 256, 0, stream>>>(XB, 1024, WT[1], 1024, sa_bk,
      nullptr, KB, 1024, (int)MQ, 1024, 1024, 0, 0, 0, 0, 0, 0, 0, 1);
  gemm_bf16_wmma<<<dim3(16, 32, 1), 256, 0, stream>>>(XB, 1024, WT[2], 1024, sa_bv,
      nullptr, VB, 1024, (int)MQ, 1024, 1024, 0, 0, 0, 0, 0, 0, 0, 1);

  // ---- 3. V^T per (b,h): [512,64] -> [64,512] ---------------------------
  transpose_b16<<<dim3(2, 16, B * H), tb, 0, stream>>>(VB, VT, 512, 64, 1024, 512,
      H, 512L * 1024, 64, 16L * 64 * 512, 64L * 512);

  // ---- 4. Self scores S = Q K^T  (f32 into attn scratch region) ---------
  gemm_bf16_wmma<<<dim3(8, 2, B * H), 256, 0, stream>>>(QB, 1024, KB, 1024, nullptr,
      attn, nullptr, 512, 512, 512, 64, 0,
      512L * 1024, 64, 512L * 1024, 64, 16L * 512 * 512, 512L * 512, H);

  // ---- 5. Causal softmax (scale 1/sqrt(64)) -> bf16 probs ---------------
  softmax_rows<<<(unsigned)(B * H * T), 256, 0, stream>>>(attn, nullptr, PB,
      T, 512, 0.125f, 1);

  // ---- 6. P V -> per-head outputs [B,T,H*64] bf16 -----------------------
  gemm_bf16_wmma<<<dim3(1, 2, B * H), 256, 0, stream>>>(PB, 512, VT, 512, nullptr,
      nullptr, AOB, 1024, 512, 64, 512, 0,
      16L * 512 * 512, 512L * 512, 16L * 64 * 512, 64L * 512, 512L * 1024, 64, H);

  // ---- 7. Output projection + LN1 ---------------------------------------
  gemm_bf16_wmma<<<dim3(16, 32, 1), 256, 0, stream>>>(AOB, 1024, WT[3], 1024, sa_bo,
      PO, nullptr, 1024, (int)MQ, 1024, 1024, 0, 0, 0, 0, 0, 0, 0, 1);
  add_layernorm<<<(unsigned)MQ, 256, 0, stream>>>(text, PO, ln1_g, ln1_b, RES, XB);

  // ---- 8. conv [B,C,1024] -> kv [B,1024,C] bf16 -------------------------
  transpose_cvt<<<dim3(32, 32, B), tb, 0, stream>>>(conv, KVB, 1024, 1024, 1024, 1024,
      1024L * 1024, 1024L * 1024);

  // ---- 9. Cross-attention projections -----------------------------------
  gemm_bf16_wmma<<<dim3(16, 32, 1), 256, 0, stream>>>(XB, 1024, WT[4], 1024, ca_bq,
      nullptr, QB, 1024, (int)MQ, 1024, 1024, 0, 0, 0, 0, 0, 0, 0, 1);
  gemm_bf16_wmma<<<dim3(16, 64, 1), 256, 0, stream>>>(KVB, 1024, WT[5], 1024, ca_bk,
      nullptr, KB, 1024, (int)MKV, 1024, 1024, 0, 0, 0, 0, 0, 0, 0, 1);
  gemm_bf16_wmma<<<dim3(16, 64, 1), 256, 0, stream>>>(KVB, 1024, WT[6], 1024, ca_bv,
      nullptr, VB, 1024, (int)MKV, 1024, 1024, 0, 0, 0, 0, 0, 0, 0, 1);

  // ---- 10. V^T per (b,h): [1024,64] -> [64,1024] ------------------------
  transpose_b16<<<dim3(2, 32, B * H), tb, 0, stream>>>(VB, VT, 1024, 64, 1024, 1024,
      H, 1024L * 1024, 64, 16L * 64 * 1024, 64L * 1024);

  // ---- 11. Cross scores (f32 directly into attn_map region) -------------
  gemm_bf16_wmma<<<dim3(16, 2, B * H), 256, 0, stream>>>(QB, 1024, KB, 1024, nullptr,
      attn, nullptr, 1024, 512, 1024, 64, 0,
      512L * 1024, 64, 1024L * 1024, 64, 16L * 512 * 1024, 512L * 1024, H);

  // ---- 12. Softmax: f32 probs in place (final attn_map) + bf16 ----------
  softmax_rows<<<(unsigned)(B * H * T), 256, 0, stream>>>(attn, attn, PB,
      T, TKC, 0.125f, 0);

  // ---- 13. P V -----------------------------------------------------------
  gemm_bf16_wmma<<<dim3(1, 2, B * H), 256, 0, stream>>>(PB, 1024, VT, 1024, nullptr,
      nullptr, AOB, 1024, 512, 64, 1024, 0,
      16L * 512 * 1024, 512L * 1024, 16L * 64 * 1024, 64L * 1024, 512L * 1024, 64, H);

  // ---- 14. Output projection + LN2 --------------------------------------
  gemm_bf16_wmma<<<dim3(16, 32, 1), 256, 0, stream>>>(AOB, 1024, WT[7], 1024, ca_bo,
      PO, nullptr, 1024, (int)MQ, 1024, 1024, 0, 0, 0, 0, 0, 0, 0, 1);
  add_layernorm<<<(unsigned)MQ, 256, 0, stream>>>(RES, PO, ln2_g, ln2_b, RES, XB);

  // ---- 15. FFN: relu(X W1 + b1) W2 + b2 ---------------------------------
  gemm_bf16_wmma<<<dim3(32, 32, 1), 256, 0, stream>>>(XB, 1024, W1T, 1024, ffn_b1,
      nullptr, HB, 2048, (int)MQ, DFF, 1024, 1, 0, 0, 0, 0, 0, 0, 1);
  gemm_bf16_wmma<<<dim3(16, 32, 1), 256, 0, stream>>>(HB, 2048, W2T, 2048, ffn_b2,
      PO, nullptr, 1024, (int)MQ, 1024, DFF, 0, 0, 0, 0, 0, 0, 0, 1);

  // ---- 16. LN3 -> final result ------------------------------------------
  add_layernorm<<<(unsigned)MQ, 256, 0, stream>>>(RES, PO, ln3_g, ln3_b, outRes, nullptr);
}
